// FlakyDetect_76012331204612
// MI455X (gfx1250) — matrix operand order repaired and checked by reading, not compiled, not run on hardware
//
#include <hip/hip_runtime.h>
#include <hip/hip_bf16.h>

// ---------------------------------------------------------------------------
// FlakyDetect pipeline for gfx1250 (MI455X), wave32 + WMMA bf16 + async-LDS.
//
//  K1 proj_ln : x = tanh(LN(inputs @ W_in^T))            -> ws.x  (bf16, 64MB)
//               (batched B-frags, LDS-staged coalesced b128 stores)
//  K2 lstm    : weights-in-VGPR LSTM, 16 rows/block      -> ws.h  (f32, 4MB)
//               (x tiles double-buffered in LDS via global_load_async_to_lds)
//  K3 seg_attn: per-group softmax attention + mean       -> ws.code (f32)
//  K4 codevec : code @ W_out^T via WMMA                  -> d_out[2048..]
//  K5 head    : code_vectors @ W_cls^T + attn_param copy -> d_out[0..], tail
// ---------------------------------------------------------------------------

#define NN   8192
#define LLEN 32
#define DIN  256
#define EE   128
#define HH   128
#define GG   1024
#define H4   512

typedef __bf16 bf16_t;
typedef __attribute__((ext_vector_type(16))) __bf16 v16bf;
typedef __attribute__((ext_vector_type(8)))  float  v8f;

__device__ __forceinline__ bf16_t f2bf(float x) { return (bf16_t)x; }

__device__ __forceinline__ v8f wmma_bf16(v16bf a, v16bf b, v8f c) {
  // (neg_a, A, neg_b, B, c_mod, C, reuse_a, reuse_b)
  return __builtin_amdgcn_wmma_f32_16x16x32_bf16(false, a, false, b,
                                                 (short)0, c, false, false);
}

__device__ __forceinline__ float sigm(float x) { return 1.0f / (1.0f + __expf(-x)); }

// Async copy of 16 bytes per lane from global to LDS (gfx1250 ASYNCcnt path).
__device__ __forceinline__ void async_b128_to_lds(const bf16_t* gsrc, bf16_t* ldst) {
  uint32_t loff = (uint32_t)(size_t)ldst;   // low 32 bits of LDS pointer = LDS offset
  uint64_t gadr = (uint64_t)(size_t)gsrc;
  asm volatile("global_load_async_to_lds_b128 %0, %1, off"
               :: "v"(loff), "v"(gadr) : "memory");
}
__device__ __forceinline__ void wait_asynccnt_le1() {
  asm volatile("s_wait_asynccnt 0x1" ::: "memory");
}

// ---------------------------------------------------------------------------
// K1: input projection + layernorm + tanh, bf16 output.
// Grid: 2048 blocks x 256 threads. Each wave: 16 rows x 128 cols, K = 256.
// ---------------------------------------------------------------------------
__global__ void __launch_bounds__(256) proj_ln_kernel(
    const float* __restrict__ inputs, const float* __restrict__ W_in,
    const float* __restrict__ gamma, const float* __restrict__ beta,
    bf16_t* __restrict__ xout) {
  __shared__ bf16_t sW[EE * DIN];  // 64KB: W_in bf16; reused later as output stage
  for (int i = threadIdx.x; i < EE * DIN; i += 256) sW[i] = f2bf(W_in[i]);
  __syncthreads();

  const int wave  = threadIdx.x >> 5;
  const int lane  = threadIdx.x & 31;
  const int l15   = lane & 15;
  const int lhalf = lane >> 4;  // 0: lanes 0-15, 1: lanes 16-31

  const size_t rowbase = (size_t)blockIdx.x * 128 + wave * 16;
  const float* ap0 = inputs + (rowbase + l15) * DIN;

  v8f acc[8];
#pragma unroll
  for (int nt = 0; nt < 8; ++nt)
    acc[nt] = (v8f){0.f, 0.f, 0.f, 0.f, 0.f, 0.f, 0.f, 0.f};

#pragma unroll
  for (int kk = 0; kk < 8; ++kk) {   // K = 256 in steps of 32
    const int K0  = kk * 32;
    const int klo = lhalf * 8;       // A-layout K interleave between halves
    v16bf a;
#pragma unroll
    for (int j = 0; j < 8; ++j) a[j]     = f2bf(ap0[K0 + klo + j]);
#pragma unroll
    for (int j = 0; j < 8; ++j) a[8 + j] = f2bf(ap0[K0 + 16 + klo + j]);

    const int kb = K0 + lhalf * 8;   // B-layout: lane n holds column n
    // Batch all 8 B fragments first, then issue 8 WMMAs back-to-back.
    v16bf bfr[8];
#pragma unroll
    for (int nt = 0; nt < 8; ++nt) {
      const bf16_t* bp = sW + (size_t)(nt * 16 + l15) * DIN + kb;
#pragma unroll
      for (int j = 0; j < 16; ++j) bfr[nt][j] = bp[(j & 7) + ((j >> 3) << 4)];
    }
#pragma unroll
    for (int nt = 0; nt < 8; ++nt) acc[nt] = wmma_bf16(a, bfr[nt], acc[nt]);
  }

  // LayerNorm over the 128 columns of each row (rows r / r+8 per lane half).
  float s1[8], s2[8];
#pragma unroll
  for (int r = 0; r < 8; ++r) {
    float a1 = 0.f, a2 = 0.f;
#pragma unroll
    for (int nt = 0; nt < 8; ++nt) { float v = acc[nt][r]; a1 += v; a2 += v * v; }
    s1[r] = a1; s2[r] = a2;
  }
#pragma unroll
  for (int r = 0; r < 8; ++r) {
#pragma unroll
    for (int off = 1; off <= 8; off <<= 1) {  // reduce within each 16-lane group
      s1[r] += __shfl_xor(s1[r], off, 32);
      s2[r] += __shfl_xor(s2[r], off, 32);
    }
  }
  float mu[8], inv[8];
#pragma unroll
  for (int r = 0; r < 8; ++r) {
    float m = s1[r] * (1.0f / 128.0f);
    float v = s2[r] * (1.0f / 128.0f) - m * m;
    mu[r] = m;
    inv[r] = rsqrtf(v + 1e-5f);
  }

  // All waves are done reading sW; reuse it as the output staging tile
  // [128 rows][128 cols] bf16 = 32KB so global stores can be coalesced b128.
  __syncthreads();
  bf16_t* sXt = sW;
#pragma unroll
  for (int nt = 0; nt < 8; ++nt) {
    const int col = nt * 16 + l15;
    const float gm = gamma[col], bt = beta[col];
#pragma unroll
    for (int r = 0; r < 8; ++r) {
      float y = tanhf((acc[nt][r] - mu[r]) * inv[r] * gm + bt);
      const int rloc = wave * 16 + r + lhalf * 8;
      sXt[rloc * EE + col] = f2bf(y);
    }
  }
  __syncthreads();

  // Coalesced writeback: block output is one contiguous 32KB range.
  uint4* g4 = (uint4*)(xout + (size_t)blockIdx.x * 128 * EE);
  const uint4* s4 = (const uint4*)sXt;
#pragma unroll
  for (int i = 0; i < 8; ++i) g4[threadIdx.x * 8 + i] = s4[threadIdx.x * 8 + i];
}

// ---------------------------------------------------------------------------
// K2: LSTM, weights resident in VGPRs, x tiles async-double-buffered in LDS.
// Grid: 512 blocks x 256 threads; each block owns 16 sequence rows.
// Per wave: 4 gate-column tiles, B-frags for Wih & Whh held in registers
// (32 x v16bf = 256 VGPRs). Gates + h + x buffers live in LDS (~30KB).
// ---------------------------------------------------------------------------
__global__ void __launch_bounds__(256) lstm_kernel(
    const bf16_t* __restrict__ x, const int* __restrict__ lengths,
    const float* __restrict__ Wih, const float* __restrict__ Whh,
    const float* __restrict__ bih, const float* __restrict__ bhh,
    float* __restrict__ hout) {
  __shared__ bf16_t sG[16 * H4];      // 16KB gate pre-activations
  __shared__ bf16_t sH[16 * HH];      //  4KB hidden state (bf16 for A-frags)
  __shared__ bf16_t sX[2][16 * HH];   //  8KB x_t double buffer (async target)
  __shared__ float  sB[H4];           //  2KB fused bias bih+bhh

  const int tid   = threadIdx.x;
  const int wave  = tid >> 5;
  const int lane  = tid & 31;
  const int l15   = lane & 15;
  const int lhalf = lane >> 4;
  const int rowbase = blockIdx.x * 16;

  for (int i = tid; i < H4; i += 256) sB[i] = bih[i] + bhh[i];
  for (int i = tid; i < 16 * HH; i += 256) sH[i] = f2bf(0.0f);

  // Per-thread slice of the 4KB x tile: 16B each (256 threads x 16B = 4KB).
  const int xrow  = tid >> 4;          // 0..15
  const int xcol8 = (tid & 15) * 8;    // 0,8,...,120

  // Kick off async prefetch of the t=0 tile.
  {
    const bf16_t* src = x + ((size_t)(rowbase + xrow) * LLEN + 0) * EE + xcol8;
    async_b128_to_lds(src, &sX[0][tid * 8]);
  }

  // Resident B fragments for this wave's 4 gate-column tiles.
  v16bf bI[4][4], bHh[4][4];
#pragma unroll
  for (int ct = 0; ct < 4; ++ct) {
    const int n = (wave * 4 + ct) * 16 + l15;  // gate column index 0..511
#pragma unroll
    for (int kk = 0; kk < 4; ++kk) {
      const int kb = kk * 32 + lhalf * 8;
      const float* pI = Wih + (size_t)n * EE + kb;
      const float* pH = Whh + (size_t)n * HH + kb;
#pragma unroll
      for (int j = 0; j < 16; ++j) {
        bI[ct][kk][j]  = f2bf(pI[(j & 7) + ((j >> 3) << 4)]);
        bHh[ct][kk][j] = f2bf(pH[(j & 7) + ((j >> 3) << 4)]);
      }
    }
  }

  // Per-thread elementwise state: row m = tid/16, cols j0..j0+7.
  const int m  = tid >> 4;
  const int j0 = (tid & 15) * 8;
  float c[8], hreg[8];
#pragma unroll
  for (int q = 0; q < 8; ++q) { c[q] = 0.f; hreg[q] = 0.f; }
  const int mylen = lengths[rowbase + m];

  __syncthreads();

  for (int t = 0; t < LLEN; ++t) {
    // Prefetch tile t+1 (wrapped at t=31: harmless refill of the dead buffer).
    {
      const int tn = (t + 1) & (LLEN - 1);
      const bf16_t* src = x + ((size_t)(rowbase + xrow) * LLEN + tn) * EE + xcol8;
      async_b128_to_lds(src, &sX[(t + 1) & 1][tid * 8]);
    }
    // Ensure tile t has landed (1 newer async op still in flight), then sync
    // so every thread's slice is visible to the whole block.
    wait_asynccnt_le1();
    __syncthreads();

    // ---- A fragments: x_t and h, both from LDS ----
    const int klo = lhalf * 8;
    const bf16_t* xp = &sX[t & 1][l15 * HH];
    const bf16_t* hp = sH + (size_t)l15 * HH;
    v16bf aX[4], aH[4];
#pragma unroll
    for (int kk = 0; kk < 4; ++kk) {
      const int K0 = kk * 32;
#pragma unroll
      for (int j = 0; j < 8; ++j) aX[kk][j]     = xp[K0 + klo + j];
#pragma unroll
      for (int j = 0; j < 8; ++j) aX[kk][8 + j] = xp[K0 + 16 + klo + j];
#pragma unroll
      for (int j = 0; j < 8; ++j) aH[kk][j]     = hp[K0 + klo + j];
#pragma unroll
      for (int j = 0; j < 8; ++j) aH[kk][8 + j] = hp[K0 + 16 + klo + j];
    }

    // ---- gates = x_t@Wih^T + h@Whh^T + b ----
#pragma unroll
    for (int ct = 0; ct < 4; ++ct) {
      const int n = (wave * 4 + ct) * 16 + l15;
      const float bias = sB[n];
      v8f acc = (v8f){bias, bias, bias, bias, bias, bias, bias, bias};
#pragma unroll
      for (int kk = 0; kk < 4; ++kk) acc = wmma_bf16(aX[kk], bI[ct][kk], acc);
#pragma unroll
      for (int kk = 0; kk < 4; ++kk) acc = wmma_bf16(aH[kk], bHh[ct][kk], acc);
#pragma unroll
      for (int r = 0; r < 8; ++r) {
        const int mm = r + lhalf * 8;
        sG[mm * H4 + n] = f2bf(acc[r]);
      }
    }
    __syncthreads();

    // ---- elementwise cell update (freeze past sequence length) ----
    if (t < mylen) {
#pragma unroll
      for (int q = 0; q < 8; ++q) {
        const int j = j0 + q;
        const float gi = (float)sG[m * H4 + j];
        const float gf = (float)sG[m * H4 + 128 + j];
        const float gg = (float)sG[m * H4 + 256 + j];
        const float go = (float)sG[m * H4 + 384 + j];
        const float cn = sigm(gf) * c[q] + sigm(gi) * tanhf(gg);
        c[q] = cn;
        hreg[q] = sigm(go) * tanhf(cn);
      }
    }
#pragma unroll
    for (int q = 0; q < 8; ++q) sH[m * HH + j0 + q] = f2bf(hreg[q]);
    __syncthreads();
  }

#pragma unroll
  for (int q = 0; q < 8; ++q)
    hout[(size_t)(rowbase + m) * HH + j0 + q] = hreg[q];
}

// ---------------------------------------------------------------------------
// K3: segmented softmax attention + mean.  One block per group (1024 blocks).
// segment_ids are sorted; binary-search the group's [start,end) range.
// ---------------------------------------------------------------------------
#define MAXG 1024
__global__ void __launch_bounds__(128) seg_attn_kernel(
    const float* __restrict__ h, const int* __restrict__ seg,
    const float* __restrict__ attn, float* __restrict__ code) {
  __shared__ int   sBound[2];
  __shared__ float sRed[128];
  __shared__ float sLogit[MAXG];

  const int g = blockIdx.x;
  if (threadIdx.x < 2) {
    const int target = g + threadIdx.x;
    int lo = 0, hi = NN;
    while (lo < hi) { int mid = (lo + hi) >> 1; if (seg[mid] < target) lo = mid + 1; else hi = mid; }
    sBound[threadIdx.x] = lo;
  }
  __syncthreads();
  const int start = sBound[0], end = sBound[1];
  int cnt = end - start;
  if (cnt > MAXG) cnt = MAXG;

  // logits = h @ attn, one row per wave iteration
  const int wave = threadIdx.x >> 5, lane = threadIdx.x & 31;
  for (int r = wave; r < cnt; r += 4) {
    const float* hr = h + (size_t)(start + r) * HH;
    float p = 0.f;
#pragma unroll
    for (int q = 0; q < 4; ++q) p += hr[lane * 4 + q] * attn[lane * 4 + q];
#pragma unroll
    for (int off = 16; off > 0; off >>= 1) p += __shfl_xor(p, off, 32);
    if (lane == 0) sLogit[r] = p;
  }
  __syncthreads();

  // group max
  float mx = -3.402823466e38f;
  for (int r = threadIdx.x; r < cnt; r += 128) mx = fmaxf(mx, sLogit[r]);
  sRed[threadIdx.x] = mx;
  __syncthreads();
  for (int s = 64; s > 0; s >>= 1) {
    if (threadIdx.x < s) sRed[threadIdx.x] = fmaxf(sRed[threadIdx.x], sRed[threadIdx.x + s]);
    __syncthreads();
  }
  mx = sRed[0];
  __syncthreads();

  // exp + group sum
  float z = 0.f;
  for (int r = threadIdx.x; r < cnt; r += 128) {
    float e = __expf(sLogit[r] - mx);
    sLogit[r] = e;
    z += e;
  }
  sRed[threadIdx.x] = z;
  __syncthreads();
  for (int s = 64; s > 0; s >>= 1) {
    if (threadIdx.x < s) sRed[threadIdx.x] += sRed[threadIdx.x + s];
    __syncthreads();
  }
  z = sRed[0];
  __syncthreads();

  // code[g][j] = (1/cnt) * sum_r softmax_r * h[r][j]
  const int j = threadIdx.x;
  float acc = 0.f;
  for (int r = 0; r < cnt; ++r)
    acc += sLogit[r] * h[(size_t)(start + r) * HH + j];
  code[(size_t)g * HH + j] = acc / (z * (float)(end - start));
}

// ---------------------------------------------------------------------------
// K4: code_vectors = code @ W_out^T via WMMA.  8 blocks x 256 threads.
// ---------------------------------------------------------------------------
__global__ void __launch_bounds__(256) codevec_kernel(
    const float* __restrict__ code, const float* __restrict__ W_out,
    float* __restrict__ cv) {
  const int wave  = threadIdx.x >> 5;
  const int lane  = threadIdx.x & 31;
  const int l15   = lane & 15;
  const int lhalf = lane >> 4;
  const int rowbase = blockIdx.x * 128 + wave * 16;

  v8f acc[8];
#pragma unroll
  for (int nt = 0; nt < 8; ++nt)
    acc[nt] = (v8f){0.f, 0.f, 0.f, 0.f, 0.f, 0.f, 0.f, 0.f};

#pragma unroll
  for (int kk = 0; kk < 4; ++kk) {
    const int K0  = kk * 32;
    const int klo = lhalf * 8;
    const float* ap = code + (size_t)(rowbase + l15) * HH + K0;
    v16bf a;
#pragma unroll
    for (int j = 0; j < 8; ++j) a[j]     = f2bf(ap[klo + j]);
#pragma unroll
    for (int j = 0; j < 8; ++j) a[8 + j] = f2bf(ap[16 + klo + j]);

    const int kb = K0 + lhalf * 8;
    v16bf bfr[8];
#pragma unroll
    for (int nt = 0; nt < 8; ++nt) {
      const float* bp = W_out + (size_t)(nt * 16 + l15) * HH + kb;
#pragma unroll
      for (int j = 0; j < 16; ++j) bfr[nt][j] = f2bf(bp[(j & 7) + ((j >> 3) << 4)]);
    }
#pragma unroll
    for (int nt = 0; nt < 8; ++nt) acc[nt] = wmma_bf16(a, bfr[nt], acc[nt]);
  }

#pragma unroll
  for (int nt = 0; nt < 8; ++nt) {
    const int tcol = nt * 16 + l15;
#pragma unroll
    for (int r = 0; r < 8; ++r) {
      const int gi = rowbase + r + lhalf * 8;
      cv[(size_t)gi * 128 + tcol] = acc[nt][r];
    }
  }
}

// ---------------------------------------------------------------------------
// K5: outputs = code_vectors @ W_cls^T  (+ copy attn_param to the tail).
// ---------------------------------------------------------------------------
__global__ void __launch_bounds__(256) head_kernel(
    const float* __restrict__ cv, const float* __restrict__ W_cls,
    const float* __restrict__ attn, float* __restrict__ out) {
  const int t = blockIdx.x * blockDim.x + threadIdx.x;  // 0..2047
  const int g = t >> 1, c = t & 1;
  const float* row = cv + (size_t)g * 128;
  const float* w   = W_cls + (size_t)c * 128;
  float acc = 0.f;
#pragma unroll 4
  for (int k = 0; k < 128; ++k) acc += row[k] * w[k];
  out[g * 2 + c] = acc;
  if (t < 128) out[2048 + 131072 + t] = attn[t];  // attn_param tail
}

// ---------------------------------------------------------------------------
extern "C" void kernel_launch(void* const* d_in, const int* in_sizes, int n_in,
                              void* d_out, int out_size, void* d_ws, size_t ws_size,
                              hipStream_t stream) {
  (void)in_sizes; (void)n_in; (void)out_size; (void)ws_size;

  const float* inputs  = (const float*)d_in[0];
  const int*   lengths = (const int*)  d_in[1];
  const int*   seg     = (const int*)  d_in[2];
  const float* W_in    = (const float*)d_in[3];
  const float* gamma   = (const float*)d_in[4];
  const float* beta    = (const float*)d_in[5];
  const float* Wih     = (const float*)d_in[6];
  const float* Whh     = (const float*)d_in[7];
  const float* bih     = (const float*)d_in[8];
  const float* bhh     = (const float*)d_in[9];
  const float* attn    = (const float*)d_in[10];
  const float* W_out   = (const float*)d_in[11];
  const float* W_cls   = (const float*)d_in[12];
  float* out = (float*)d_out;

  // workspace: x_bf16 (64MB) | h (4MB) | code (512KB)  -> ~68.6MB total
  char* ws = (char*)d_ws;
  bf16_t* x_bf = (bf16_t*)ws;
  float*  h_ws = (float*)(ws + (size_t)NN * LLEN * EE * sizeof(bf16_t));
  float*  codw = (float*)((char*)h_ws + (size_t)NN * HH * sizeof(float));
  float*  cv   = out + 2048;  // code_vectors region of d_out

  proj_ln_kernel<<<dim3(2048), dim3(256), 0, stream>>>(inputs, W_in, gamma, beta, x_bf);
  lstm_kernel<<<dim3(512), dim3(256), 0, stream>>>(x_bf, lengths, Wih, Whh, bih, bhh, h_ws);
  seg_attn_kernel<<<dim3(1024), dim3(128), 0, stream>>>(h_ws, seg, attn, codw);
  codevec_kernel<<<dim3(8), dim3(256), 0, stream>>>(codw, W_out, cv);
  head_kernel<<<dim3(8), dim3(256), 0, stream>>>(cv, W_cls, attn, out);
}
